// SymmetryConstraint_63745904608093
// MI455X (gfx1250) — matrix operand order repaired and checked by reading, not compiled, not run on hardware
//
#include <hip/hip_runtime.h>

typedef __attribute__((ext_vector_type(16))) _Float16 v16h;
typedef __attribute__((ext_vector_type(8)))  float    v8f;

#define NPTS 512          // keypoints per batch (N)
#define CENTER_X 0.5f

// One wave (32 lanes) per batch. Computes, via f16 WMMA with f32 accumulation,
// the 5 masked reductions per symmetric class c in {0,1,2}:
//   n_c = sum(m), U_c = sum(m*u), Y_c = sum(m*y), SU_c = sum(m*u^2), SY_c = sum(m*y^2)
// with u = x - CENTER_X, m = (class == c). Then the closed-form pairwise loss:
//   S_c = (n_c-1)(SU_c+SY_c) + U_c^2 - SU_c - Y_c^2 + SY_c
//   cnt_c = n_c(n_c-1)/2
__global__ __launch_bounds__(32) void symc_batch_kernel(
    const float* __restrict__ kp,    // [B, N, 2] float32
    const int*   __restrict__ cls,   // [B, N]    int32
    float* __restrict__ partial)     // [B*2]: (S_b, cnt_b)
{
    __shared__ float s_u[NPTS];
    __shared__ float s_y[NPTS];
    __shared__ int   s_c[NPTS];

    const int b    = blockIdx.x;
    const int lane = threadIdx.x;            // 0..31 (wave32)

    const float* kpb = kp  + (size_t)b * NPTS * 2;
    const int*   cb  = cls + (size_t)b * NPTS;

    // Stage this batch into LDS (8B per-lane vector loads via float2 pairs).
    for (int i = lane; i < NPTS; i += 32) {
        const float2 p = ((const float2*)kpb)[i];
        s_u[i] = p.x - CENTER_X;
        s_y[i] = p.y;
        s_c[i] = cb[i];
    }
    __syncthreads();

    // A (16x32 f16): row q = lane%16 carries data type q: {1, u, y, u^2, y^2}, rows 5..15 = 0.
//   element e <-> K = e + 16*(lane>=16)                        (same K order as B)
    // B (32x16 f16): col n = lane%16 is one-hot mask for class n (cols 3..15 = 0).
    //   element e <-> K = e + 16*(lane>=16)                        (same K order as A)
    const int q  = lane & 15;
    const int hA = lane >> 4;
    const int n  = lane & 15;
    const int hB = lane >> 4;

    // Branch-free per-lane row selectors (one-hot weights), computed once.
    const float w0 = (q == 0) ? 1.0f : 0.0f;
    const float w1 = (q == 1) ? 1.0f : 0.0f;
    const float w2 = (q == 2) ? 1.0f : 0.0f;
    const float w3 = (q == 3) ? 1.0f : 0.0f;
    const float w4 = (q == 4) ? 1.0f : 0.0f;
    // Per-lane mask target: class n for columns 0..2, otherwise a value that
    // can never match (classes are in [-2, 6)).
    const int ctarget = (n < 3) ? n : 0x7fffffff;

    v8f acc = {};
#pragma unroll 4
    for (int t = 0; t < NPTS / 32; ++t) {
        const int base = t * 32;
        v16h afrag, bfrag;
#pragma unroll
        for (int e = 0; e < 16; ++e) {
            const int ka = base + e + 16 * hA;
            const float uk = s_u[ka];
            const float yk = s_y[ka];
            // av = w0*1 + w1*u + w2*y + w3*u^2 + w4*y^2  (pure FMA select)
            const float av = w0 + uk * (w1 + w3 * uk) + yk * (w2 + w4 * yk);
            afrag[e] = (_Float16)av;

            const int kb = base + e + 16 * hB;
            bfrag[e] = (s_c[kb] == ctarget) ? (_Float16)1.0f : (_Float16)0.0f;
        }
        // D = A x B + C ; emits v_wmma_f32_16x16x32_f16 (f32 accumulation)
        acc = __builtin_amdgcn_wmma_f32_16x16x32_f16(
            /*neg_a=*/false, afrag, /*neg_b=*/false, bfrag,
            /*c_mod=*/(short)0, acc, /*reuse_a=*/false, /*reuse_b=*/false);
    }

    // C layout: lane n (0..15) holds column n; VGPR r holds row r (lanes 0..15).
    // => lanes 0..2 hold, for class c = lane: acc[0]=n_c, acc[1]=U_c, acc[2]=Y_c,
    //    acc[3]=SU_c, acc[4]=SY_c.
    const float nc = acc[0], U = acc[1], Y = acc[2], SU = acc[3], SY = acc[4];
    float S   = (nc - 1.0f) * (SU + SY) + U * U - SU - Y * Y + SY;
    float cnt = 0.5f * nc * (nc - 1.0f);
    if (lane >= 3) { S = 0.0f; cnt = 0.0f; }

    // Lanes 0..2 live in the xor-group {0,1,2,3}; two xor-shuffles reduce them.
    S   += __shfl_xor(S,   1, 32);
    S   += __shfl_xor(S,   2, 32);
    cnt += __shfl_xor(cnt, 1, 32);
    cnt += __shfl_xor(cnt, 2, 32);

    if (lane == 0) {
        partial[2 * b + 0] = S;
        partial[2 * b + 1] = cnt;
    }
}

// Reduce the per-batch partials and apply the final division.
__global__ __launch_bounds__(256) void symc_final_kernel(
    const float* __restrict__ partial, float* __restrict__ out, int nbatch)
{
    __shared__ float sS[256];
    __shared__ float sC[256];
    const int t = threadIdx.x;
    float S = 0.0f, C = 0.0f;
    for (int i = t; i < nbatch; i += 256) {
        S += partial[2 * i + 0];
        C += partial[2 * i + 1];
    }
    sS[t] = S;
    sC[t] = C;
    __syncthreads();
    for (int s = 128; s > 0; s >>= 1) {
        if (t < s) { sS[t] += sS[t + s]; sC[t] += sC[t + s]; }
        __syncthreads();
    }
    if (t == 0) out[0] = sS[0] / fmaxf(sC[0], 1.0f);
}

extern "C" void kernel_launch(void* const* d_in, const int* in_sizes, int n_in,
                              void* d_out, int out_size, void* d_ws, size_t ws_size,
                              hipStream_t stream)
{
    const float* kp  = (const float*)d_in[0];   // predicted_keypoints [B, N, 2] f32
    const int*   cls = (const int*)d_in[1];     // keypoint_classes   [B, N]    i32
    const int    B   = in_sizes[1] / NPTS;      // 256

    float* partial = (float*)d_ws;              // B*2 floats of scratch

    symc_batch_kernel<<<B, 32, 0, stream>>>(kp, cls, partial);
    symc_final_kernel<<<1, 256, 0, stream>>>(partial, (float*)d_out, B);
}
